// SegFormerXAttention_19069654794255
// MI455X (gfx1250) — matrix operand order repaired and checked by reading, not compiled, not run on hardware
//
#include <hip/hip_runtime.h>

// SegFormerX cross-attention for MI455X (gfx1250, wave32, WMMA bf16 + async LDS DMA).
// B=2, LV=2048, LT=256, D=512, H=8, DH=64. Masks in setup_inputs are all-true.

#define BB 2
#define LVV 2048
#define LTT 256
#define DD 512
#define HH 8
#define DHH 64
#define LK_TOT (LVV + LTT)

typedef __attribute__((ext_vector_type(16))) __bf16 v16bf;
typedef __attribute__((ext_vector_type(8)))  __bf16 v8bf;
typedef __attribute__((ext_vector_type(8)))  float  v8f;

union V16U { v16bf v; v8bf h[2]; unsigned short u[16]; };

__device__ __forceinline__ unsigned short f2bf(float f) {
  unsigned u = __builtin_bit_cast(unsigned, f);
  u += 0x7FFFu + ((u >> 16) & 1u);   // round-to-nearest-even
  return (unsigned short)(u >> 16);
}

// A-operand (16x32 bf16, M x K): lane row m = lane&15, khalf = lane>>4.
// Lane holds K = k0+khalf*8+{0..7} then K = k0+16+khalf*8+{0..7}.
__device__ __forceinline__ v16bf load_fragA(const unsigned short* X, int ld, int m,
                                            int k0, int khalf) {
  const unsigned short* r = X + (size_t)m * ld + k0 + khalf * 8;
  V16U u;
  u.h[0] = *reinterpret_cast<const v8bf*>(r);
  u.h[1] = *reinterpret_cast<const v8bf*>(r + 16);
  return u.v;
}

// B-operand (32x16 bf16, K x N) for Y = X * W^T style: lane col n, khalf = lane>>4.
// Lane holds K = k0+khalf*8+{0..7} then K = k0+16+khalf*8+{0..7} of row n.
__device__ __forceinline__ v16bf load_fragB(const unsigned short* W, int ld, int n,
                                            int k0, int khalf) {
  const unsigned short* r = W + (size_t)n * ld + k0 + khalf * 8;
  V16U u;
  u.h[0] = *reinterpret_cast<const v8bf*>(r);
  u.h[1] = *reinterpret_cast<const v8bf*>(r + 16);
  return u.v;
}

#define WMMA_BF16(a, b, c) \
  __builtin_amdgcn_wmma_f32_16x16x32_bf16(false, (a), false, (b), (short)0, (c), false, false)

// ---------------------------------------------------------------- cvt fp32->bf16
__global__ __launch_bounds__(256)
void cvt_kernel(const float* __restrict__ in, unsigned short* __restrict__ out, int n) {
  for (int i = blockIdx.x * blockDim.x + threadIdx.x; i < n; i += gridDim.x * blockDim.x)
    out[i] = f2bf(in[i]);
}

// ---------------------------------------------------------------- GEMM  Y = X*W^T + b
// X: (M,K) bf16 row-major; W: (N,K) bf16 row-major (torch Linear layout).
// One wave per 64x16 output tile (4 M-subtiles reuse one B fragment -> 4x fewer
// weight reloads); 8 waves per block span 8 N-tiles. M must be a multiple of 64.
__global__ __launch_bounds__(256)
void gemm_bf16_kernel(const unsigned short* __restrict__ X,
                      const unsigned short* __restrict__ W,
                      const float* __restrict__ bias,
                      unsigned short* __restrict__ out16,
                      float* __restrict__ outF,
                      int M, int N, int K) {
  const int lane  = threadIdx.x & 31;
  const int wave  = threadIdx.x >> 5;
  const int tm    = blockIdx.x;             // 64-row tile
  const int tn    = blockIdx.y * 8 + wave;  // 16-col tile
  if (tn * 16 >= N) return;
  const int nl    = lane & 15;
  const int khalf = lane >> 4;
  const int n     = tn * 16 + nl;
  const int m0    = tm * 64 + nl;

  v8f c[4];
  for (int s = 0; s < 4; ++s)
    for (int i = 0; i < 8; ++i) c[s][i] = 0.f;

  for (int k0 = 0; k0 < K; k0 += 32) {
    const v16bf b = load_fragB(W, K, n, k0, khalf);
#pragma unroll
    for (int s = 0; s < 4; ++s) {
      const v16bf a = load_fragA(X, K, m0 + s * 16, k0, khalf);
      c[s] = WMMA_BF16(a, b, c[s]);
    }
  }
  const float bv = bias[n];
#pragma unroll
  for (int s = 0; s < 4; ++s) {
    const int row0 = tm * 64 + s * 16 + khalf * 8;
    if (outF) {
      for (int i = 0; i < 8; ++i) outF[(size_t)(row0 + i) * N + n] = c[s][i] + bv;
    } else {
      for (int i = 0; i < 8; ++i) out16[(size_t)(row0 + i) * N + n] = f2bf(c[s][i] + bv);
    }
  }
}

// ---------------------------------------------------------------- flash attention
// Keys = [LV vid keys | LT txt keys]; Q projection differs per key segment
// (Qa vs Qb). Per wave: one 16-row q tile, online softmax over 2304 keys in
// 32-key blocks. V tiles are double-buffered in LDS via async DMA
// (global_load_async_to_lds_b128 / ASYNCcnt) overlapping the QK^T WMMAs.
__global__ __launch_bounds__(128)
void attn_kernel(const unsigned short* __restrict__ Qa, const unsigned short* __restrict__ Ka,
                 const unsigned short* __restrict__ Qb, const unsigned short* __restrict__ Kb,
                 const unsigned short* __restrict__ Va, const unsigned short* __restrict__ Vb,
                 unsigned short* __restrict__ outp, int LQ) {
  __shared__ __align__(16) unsigned short Plds[4][16][32];      // per-wave P tile
  __shared__ __align__(16) unsigned short Vlds[4][2][32][64];   // per-wave dbl-buf V tile

  const int lane  = threadIdx.x & 31;
  const int w     = threadIdx.x >> 5;
  const int qt    = blockIdx.x * 4 + w;
  const int h     = blockIdx.y;
  const int b     = blockIdx.z;
  const int nl    = lane & 15;
  const int khalf = lane >> 4;
  const int hoff  = h * DHH;
  const int m     = qt * 16 + nl;

  const unsigned short* Qab = Qa + (size_t)b * LQ  * DD + hoff;
  const unsigned short* Qbb = Qb + (size_t)b * LQ  * DD + hoff;
  const unsigned short* Kab = Ka + (size_t)b * LVV * DD + hoff;
  const unsigned short* Kbb = Kb + (size_t)b * LTT * DD + hoff;
  const unsigned short* Vab = Va + (size_t)b * LVV * DD + hoff;
  const unsigned short* Vbb = Vb + (size_t)b * LTT * DD + hoff;

  const v16bf qa0 = load_fragA(Qab, DD, m, 0,  khalf);
  const v16bf qa1 = load_fragA(Qab, DD, m, 32, khalf);
  const v16bf qb0 = load_fragA(Qbb, DD, m, 0,  khalf);
  const v16bf qb1 = load_fragA(Qbb, DD, m, 32, khalf);

  // Async DMA of one 32x64 bf16 V tile (4KB/wave): lane owns one key row (128B),
  // eight b128 transfers; the same inst offset advances global and LDS addresses.
  auto issueV = [&](int kk, int buf) {
    const int kr = kk + lane;
    const unsigned short* vsrc = (kk < LVV) ? (Vab + (size_t)kr * DD)
                                            : (Vbb + (size_t)(kr - LVV) * DD);
    const unsigned lds = (unsigned)(unsigned long long)(const void*)&Vlds[w][buf][lane][0];
    const unsigned long long ga = (unsigned long long)vsrc;
    asm volatile(
        "global_load_async_to_lds_b128 %0, %1, off\n\t"
        "global_load_async_to_lds_b128 %0, %1, off offset:16\n\t"
        "global_load_async_to_lds_b128 %0, %1, off offset:32\n\t"
        "global_load_async_to_lds_b128 %0, %1, off offset:48\n\t"
        "global_load_async_to_lds_b128 %0, %1, off offset:64\n\t"
        "global_load_async_to_lds_b128 %0, %1, off offset:80\n\t"
        "global_load_async_to_lds_b128 %0, %1, off offset:96\n\t"
        "global_load_async_to_lds_b128 %0, %1, off offset:112"
        :: "v"(lds), "v"(ga) : "memory");
  };

  float rmax[8], rsum[8];
  v8f acc[4];
  for (int j = 0; j < 4; ++j)
    for (int i = 0; i < 8; ++i) acc[j][i] = 0.f;
  for (int i = 0; i < 8; ++i) { rmax[i] = -1e30f; rsum[i] = 0.f; }

  issueV(0, 0);                                        // prime buffer 0

  for (int kk = 0; kk < LK_TOT; kk += 32) {            // LVV%32==0: no straddle
    const int  buf  = (kk >> 5) & 1;
    const bool more = (kk + 32 < LK_TOT);
    if (more) issueV(kk + 32, buf ^ 1);                // overlap DMA with QK^T

    // ---- logits for 32 keys: two 16x16 tiles, DH=64 -> 2 WMMAs each
    const bool invid = (kk < LVV);
    const unsigned short* Kp = invid ? Kab : Kbb;
    const int kb = invid ? kk : (kk - LVV);
    v8f c0, c1;
    for (int i = 0; i < 8; ++i) { c0[i] = 0.f; c1[i] = 0.f; }
    {
      const v16bf k00 = load_fragB(Kp, DD, kb + nl,      0,  khalf);
      const v16bf k01 = load_fragB(Kp, DD, kb + nl,      32, khalf);
      const v16bf k10 = load_fragB(Kp, DD, kb + 16 + nl, 0,  khalf);
      const v16bf k11 = load_fragB(Kp, DD, kb + 16 + nl, 32, khalf);
      if (invid) {
        c0 = WMMA_BF16(qa0, k00, c0); c0 = WMMA_BF16(qa1, k01, c0);
        c1 = WMMA_BF16(qa0, k10, c1); c1 = WMMA_BF16(qa1, k11, c1);
      } else {
        c0 = WMMA_BF16(qb0, k00, c0); c0 = WMMA_BF16(qb1, k01, c0);
        c1 = WMMA_BF16(qb0, k10, c1); c1 = WMMA_BF16(qb1, k11, c1);
      }
    }
    // ---- online softmax over this 32-key block (row = khalf*8 + i)
    for (int i = 0; i < 8; ++i) {
      float v0 = c0[i] * 0.125f;                       // scale = 1/sqrt(64)
      float v1 = c1[i] * 0.125f;
      float tmax = fmaxf(v0, v1);
      for (int off = 1; off < 16; off <<= 1) tmax = fmaxf(tmax, __shfl_xor(tmax, off, 16));
      const float nm = fmaxf(rmax[i], tmax);
      const float p0 = __expf(v0 - nm);
      const float p1 = __expf(v1 - nm);
      float psum = p0 + p1;
      for (int off = 1; off < 16; off <<= 1) psum += __shfl_xor(psum, off, 16);
      const float f = __expf(rmax[i] - nm);
      rsum[i] = rsum[i] * f + psum;
      rmax[i] = nm;
      for (int j = 0; j < 4; ++j) acc[j][i] *= f;
      const int row = khalf * 8 + i;
      Plds[w][row][nl]      = f2bf(p0);
      Plds[w][row][16 + nl] = f2bf(p1);
    }
    // ---- wait for current V tile (8 older asyncs done; next tile stays in flight)
    if (more) asm volatile("s_wait_asynccnt 0x8" ::: "memory");
    else      asm volatile("s_wait_asynccnt 0x0" ::: "memory");

    // ---- O += P(16x32) * V(32x64): P in A layout, V gathered in B layout
    V16U pu;
    pu.h[0] = *reinterpret_cast<const v8bf*>(&Plds[w][nl][khalf * 16]);
    pu.h[1] = *reinterpret_cast<const v8bf*>(&Plds[w][nl][khalf * 16 + 8]);
#pragma unroll
    for (int j = 0; j < 4; ++j) {
      V16U vb;
      for (int jj = 0; jj < 16; ++jj) vb.u[jj] = Vlds[w][buf][khalf * 16 + jj][j * 16 + nl];
      acc[j] = WMMA_BF16(pu.v, vb.v, acc[j]);
    }
  }
  // ---- epilogue: normalize and store bf16 (B, LQ, D)
  const int row0 = qt * 16 + khalf * 8;
  for (int j = 0; j < 4; ++j)
    for (int i = 0; i < 8; ++i) {
      const float o = acc[j][i] / rsum[i];
      outp[((size_t)b * LQ + row0 + i) * DD + hoff + j * 16 + nl] = f2bf(o);
    }
}

// ---------------------------------------------------------------- LayerNorm(y + res)
__global__ __launch_bounds__(256)
void ln_kernel(const float* __restrict__ y, const float* __restrict__ res,
               const float* __restrict__ g, const float* __restrict__ be,
               float* __restrict__ out) {
  __shared__ float red[256];
  const int row = blockIdx.x;
  const int t   = threadIdx.x;
  const size_t base = (size_t)row * DD;
  const float v0 = y[base + t]       + res[base + t];
  const float v1 = y[base + t + 256] + res[base + t + 256];
  red[t] = v0 + v1;
  __syncthreads();
  for (int off = 128; off > 0; off >>= 1) { if (t < off) red[t] += red[t + off]; __syncthreads(); }
  const float mu = red[0] * (1.f / DD);
  __syncthreads();
  const float d0 = v0 - mu, d1 = v1 - mu;
  red[t] = d0 * d0 + d1 * d1;
  __syncthreads();
  for (int off = 128; off > 0; off >>= 1) { if (t < off) red[t] += red[t + off]; __syncthreads(); }
  const float rstd = rsqrtf(red[0] * (1.f / DD) + 1e-12f);
  out[base + t]       = d0 * rstd * g[t]       + be[t];
  out[base + t + 256] = d1 * rstd * g[t + 256] + be[t + 256];
}

// ---------------------------------------------------------------- host
extern "C" void kernel_launch(void* const* d_in, const int* in_sizes, int n_in,
                              void* d_out, int out_size, void* d_ws, size_t ws_size,
                              hipStream_t stream) {
  (void)in_sizes; (void)n_in; (void)out_size; (void)ws_size;
  const float* vid_feat = (const float*)d_in[0];
  const float* txt_feat = (const float*)d_in[2];
  const float* t2v_w = (const float*)d_in[4];  const float* t2v_b = (const float*)d_in[5];
  const float* v2v_w = (const float*)d_in[6];  const float* v2v_b = (const float*)d_in[7];
  const float* t2t_w = (const float*)d_in[8];  const float* t2t_b = (const float*)d_in[9];
  const float* v2t_w = (const float*)d_in[10]; const float* v2t_b = (const float*)d_in[11];
  const float* ff_vid_w = (const float*)d_in[12]; const float* ff_vid_b = (const float*)d_in[13];
  const float* ff_txt_w = (const float*)d_in[14]; const float* ff_txt_b = (const float*)d_in[15];
  const float* ln_vid_g = (const float*)d_in[16]; const float* ln_vid_b = (const float*)d_in[17];
  const float* ln_txt_g = (const float*)d_in[18]; const float* ln_txt_b = (const float*)d_in[19];

  const size_t NV = (size_t)BB * LVV * DD;   // 2,097,152
  const size_t NT = (size_t)BB * LTT * DD;   //   262,144
  const size_t NW = (size_t)DD * DD;         //   262,144

  char* p = (char*)d_ws;
  auto take = [&](size_t bytes) { char* r = p; p += (bytes + 255) & ~(size_t)255; return r; };

  unsigned short* XV16  = (unsigned short*)take(NV * 2);
  unsigned short* XT16  = (unsigned short*)take(NT * 2);
  unsigned short* W_t2v = (unsigned short*)take(3 * NW * 2);
  unsigned short* W_v2v = (unsigned short*)take(3 * NW * 2);
  unsigned short* W_t2t = (unsigned short*)take(3 * NW * 2);
  unsigned short* W_v2t = (unsigned short*)take(3 * NW * 2);
  unsigned short* W_ffv = (unsigned short*)take(NW * 2);
  unsigned short* W_fft = (unsigned short*)take(NW * 2);
  // projections (B, L, D) bf16
  unsigned short* Qvv = (unsigned short*)take(NV * 2);
  unsigned short* Kvv = (unsigned short*)take(NV * 2);
  unsigned short* Vvv = (unsigned short*)take(NV * 2);
  unsigned short* Qvt = (unsigned short*)take(NV * 2);
  unsigned short* Kvt = (unsigned short*)take(NV * 2);
  unsigned short* Vvt = (unsigned short*)take(NV * 2);
  unsigned short* Ktv = (unsigned short*)take(NT * 2);
  unsigned short* Vtv = (unsigned short*)take(NT * 2);
  unsigned short* Qtv = (unsigned short*)take(NT * 2);
  unsigned short* Qtt = (unsigned short*)take(NT * 2);
  unsigned short* Ktt = (unsigned short*)take(NT * 2);
  unsigned short* Vtt = (unsigned short*)take(NT * 2);
  unsigned short* AttnV = (unsigned short*)take(NV * 2);
  unsigned short* AttnT = (unsigned short*)take(NT * 2);
  float* FFV = (float*)take(NV * 4);
  float* FFT = (float*)take(NT * 4);

  // ---- fp32 -> bf16 conversions
  cvt_kernel<<<1024, 256, 0, stream>>>(vid_feat, XV16, (int)NV);
  cvt_kernel<<<256,  256, 0, stream>>>(txt_feat, XT16, (int)NT);
  cvt_kernel<<<512,  256, 0, stream>>>(t2v_w, W_t2v, (int)(3 * NW));
  cvt_kernel<<<512,  256, 0, stream>>>(v2v_w, W_v2v, (int)(3 * NW));
  cvt_kernel<<<512,  256, 0, stream>>>(t2t_w, W_t2t, (int)(3 * NW));
  cvt_kernel<<<512,  256, 0, stream>>>(v2t_w, W_v2t, (int)(3 * NW));
  cvt_kernel<<<256,  256, 0, stream>>>(ff_vid_w, W_ffv, (int)NW);
  cvt_kernel<<<256,  256, 0, stream>>>(ff_txt_w, W_fft, (int)NW);

  const int MV = BB * LVV;  // 4096 rows
  const int MT = BB * LTT;  //  512 rows
  const dim3 gV(MV / 64, (DD / 16) / 8);
  const dim3 gT(MT / 64, (DD / 16) / 8);

  // ---- 12 projection GEMMs (bf16 out)
  gemm_bf16_kernel<<<gV, 256, 0, stream>>>(XV16, W_v2v + 0 * NW, v2v_b + 0 * DD, Qvv, nullptr, MV, DD, DD);
  gemm_bf16_kernel<<<gV, 256, 0, stream>>>(XV16, W_v2v + 1 * NW, v2v_b + 1 * DD, Kvv, nullptr, MV, DD, DD);
  gemm_bf16_kernel<<<gV, 256, 0, stream>>>(XV16, W_v2v + 2 * NW, v2v_b + 2 * DD, Vvv, nullptr, MV, DD, DD);
  gemm_bf16_kernel<<<gV, 256, 0, stream>>>(XV16, W_t2v + 0 * NW, t2v_b + 0 * DD, Qvt, nullptr, MV, DD, DD);
  gemm_bf16_kernel<<<gV, 256, 0, stream>>>(XV16, W_v2t + 1 * NW, v2t_b + 1 * DD, Kvt, nullptr, MV, DD, DD);
  gemm_bf16_kernel<<<gV, 256, 0, stream>>>(XV16, W_v2t + 2 * NW, v2t_b + 2 * DD, Vvt, nullptr, MV, DD, DD);
  gemm_bf16_kernel<<<gT, 256, 0, stream>>>(XT16, W_t2v + 1 * NW, t2v_b + 1 * DD, Ktv, nullptr, MT, DD, DD);
  gemm_bf16_kernel<<<gT, 256, 0, stream>>>(XT16, W_t2v + 2 * NW, t2v_b + 2 * DD, Vtv, nullptr, MT, DD, DD);
  gemm_bf16_kernel<<<gT, 256, 0, stream>>>(XT16, W_v2t + 0 * NW, v2t_b + 0 * DD, Qtv, nullptr, MT, DD, DD);
  gemm_bf16_kernel<<<gT, 256, 0, stream>>>(XT16, W_t2t + 0 * NW, t2t_b + 0 * DD, Qtt, nullptr, MT, DD, DD);
  gemm_bf16_kernel<<<gT, 256, 0, stream>>>(XT16, W_t2t + 1 * NW, t2t_b + 1 * DD, Ktt, nullptr, MT, DD, DD);
  gemm_bf16_kernel<<<gT, 256, 0, stream>>>(XT16, W_t2t + 2 * NW, t2t_b + 2 * DD, Vtt, nullptr, MT, DD, DD);

  // ---- fused flash attention (video target, text target)
  attn_kernel<<<dim3(LVV / 64, HH, BB), 128, 0, stream>>>(Qvv, Kvv, Qvt, Ktv, Vvv, Vtv, AttnV, LVV);
  attn_kernel<<<dim3(LTT / 64, HH, BB), 128, 0, stream>>>(Qtv, Kvt, Qtt, Ktt, Vvt, Vtt, AttnT, LTT);

  // ---- FF GEMMs (f32 out)
  gemm_bf16_kernel<<<gV, 256, 0, stream>>>(AttnV, W_ffv, ff_vid_b, nullptr, FFV, MV, DD, DD);
  gemm_bf16_kernel<<<gT, 256, 0, stream>>>(AttnT, W_fft, ff_txt_b, nullptr, FFT, MT, DD, DD);

  // ---- residual + LayerNorm -> d_out (vid_out first, then txt_out)
  float* outv = (float*)d_out;
  float* outt = outv + NV;
  ln_kernel<<<MV, 256, 0, stream>>>(FFV, vid_feat, ln_vid_g, ln_vid_b, outv);
  ln_kernel<<<MT, 256, 0, stream>>>(FFT, txt_feat, ln_txt_g, ln_txt_b, outt);
}
